// MultiLoraRowParallelLinear_34007551050179
// MI455X (gfx1250) — matrix operand order repaired and checked by reading, not compiled, not run on hardware
//
#include <hip/hip_runtime.h>

typedef __bf16 bf16_t;
typedef __attribute__((ext_vector_type(16))) __bf16 v16bf;
typedef __attribute__((ext_vector_type(8)))  __bf16 v8bf;
typedef __attribute__((ext_vector_type(8)))  float  v8f;
typedef __attribute__((ext_vector_type(4)))  float  v4f;
typedef __attribute__((ext_vector_type(4)))  unsigned int v4u;
typedef __attribute__((ext_vector_type(8)))  int    v8i;
typedef __attribute__((ext_vector_type(4)))  int    v4i;

#define TILE_M 128
#define TILE_N 128
#define TILE_K 32
#define LDSTR  40                      // 32 + 8 bf16 pad per row (80B stride)
#define K_DIM  4096
#define N_DIM  4096
#define NTILES (K_DIM / TILE_K)        // 128
#define PLANE_E (TILE_M * LDSTR)       // 5120 bf16 per plane
#define BUF_B   (4 * PLANE_E * 2)      // Ah,Al,Bh,Bl = 40960 bytes per plane buffer
#define STAGE_A (TILE_M * TILE_K * 4)  // 16384 B f32 A tile [128][32]
#define STAGE_BUF (2 * STAGE_A)        // + f32 B tile [32][128] = 32768 B

#if __has_builtin(__builtin_amdgcn_tensor_load_to_lds)
#define HAVE_TDM 1
#else
#define HAVE_TDM 0
#endif

static __device__ __forceinline__ v16bf cat8(v8bf a, v8bf b) {
  return __builtin_shufflevector(a, b, 0,1,2,3,4,5,6,7,8,9,10,11,12,13,14,15);
}

// Exact truncation split: f == hi + lo_exact, lo rounded to bf16
static __device__ __forceinline__ void split_bf16(float f, bf16_t& h, bf16_t& l) {
  unsigned u  = __float_as_uint(f);
  float    hf = __uint_as_float(u & 0xffff0000u);
  h = __builtin_bit_cast(bf16_t, (unsigned short)(u >> 16));
  l = (bf16_t)(f - hf);
}

// s_wait_tensorcnt needs a constant immediate -> template parameter
template <short N>
static __device__ __forceinline__ void wait_fetch(int wave) {
#if HAVE_TDM
  if (wave == 0) __builtin_amdgcn_s_wait_tensorcnt(N);
#endif
}

#if HAVE_TDM
// 2D TDM load: tile (tile_d0 x tile_d1), dim0 contiguous, dim1 stride0 elems, 4B data
static __device__ __forceinline__ void tdm_load_2d(unsigned lds_off,
    unsigned long long gaddr, int tensor_d0, int tensor_d1,
    int tile_d0, int tile_d1, int stride0)
{
  v4u g0;
  g0.x = 1u;                                               // count=1, user mode
  g0.y = lds_off;                                          // lds_addr (bytes)
  g0.z = (unsigned)gaddr;                                  // global_addr[31:0]
  g0.w = (unsigned)((gaddr >> 32) & 0x01FFFFFFull) | 0x80000000u; // addr[56:32]|type=2
  v8i g1;
  g1[0] = 0x00020000;                                      // data_size=2 -> 4 bytes
  g1[1] = (tensor_d0 & 0xFFFF) << 16;                      // tensor_dim0[15:0]
  g1[2] = ((tensor_d0 >> 16) & 0xFFFF) | ((tensor_d1 & 0xFFFF) << 16);
  g1[3] = ((tensor_d1 >> 16) & 0xFFFF) | ((tile_d0 & 0xFFFF) << 16);
  g1[4] = tile_d1 & 0xFFFF;                                // tile_dim1, tile_dim2=0
  g1[5] = stride0;                                         // tensor_dim0_stride[31:0]
  g1[6] = 0;                                               // stride hi, dim1_stride lo
  g1[7] = 0;
  v4i z4 = {0,0,0,0};
#if defined(__clang_major__) && __clang_major__ >= 23
  v8i z8 = {0,0,0,0,0,0,0,0};
  __builtin_amdgcn_tensor_load_to_lds(g0, g1, z4, z4, z8, 0);
#else
  __builtin_amdgcn_tensor_load_to_lds(g0, g1, z4, z4, 0);
#endif
}
#endif

__global__ __launch_bounds__(256)
void multilora_gemm_bf16x3(const float* __restrict__ x,
                           const int*   __restrict__ adapter_ids,
                           const float* __restrict__ weight,
                           float* __restrict__ out)
{
  // [0, 2*BUF_B)            : bf16 hi/lo planes, double buffered
  // [2*BUF_B, +2*STAGE_BUF) : raw f32 tiles (TDM destination), double buffered
  __shared__ __align__(16) unsigned char smem[2 * BUF_B + 2 * STAGE_BUF]; // 147456 B
  unsigned char* lds_planes = smem;
  unsigned char* lds_stage  = smem + 2 * BUF_B;

  const int tid = threadIdx.x;
  const int bid = blockIdx.x;
  const int b   = bid >> 8;           // 256 tiles per batch
  const int t   = bid & 255;
  const int mt  = t >> 5;             // 0..7
  const int nt  = t & 31;             // 0..31

  const int aid = adapter_ids[b] & 7;
  const float* Abase = x      + (long long)(b * 1024 + mt * TILE_M) * K_DIM;
  const float* Bbase = weight + (long long)aid * K_DIM * N_DIM + (long long)nt * TILE_N;
  float*       Cbase = out    + (long long)(b * 1024 + mt * TILE_M) * N_DIM + nt * TILE_N;

  const int wave  = tid >> 5;
  const int lane  = tid & 31;
  const int wm    = wave >> 1;
  const int wn    = wave & 1;
  const int lhalf = lane & 15;
  const int lhi   = lane >> 4;

  const v8f vzero = {0.f,0.f,0.f,0.f,0.f,0.f,0.f,0.f};
  v8f acc[2][4];
  #pragma unroll
  for (int i = 0; i < 2; ++i)
    #pragma unroll
    for (int j = 0; j < 4; ++j) acc[i][j] = vzero;

  // ---- fetch tile tt (f32) into stage[tt&1] ----
  auto fetch_tile = [&](int tt) {
#if HAVE_TDM
    if (wave == 0) {
      unsigned sa = (unsigned)(unsigned long long)(lds_stage + (tt & 1) * STAGE_BUF);
      unsigned long long ga = (unsigned long long)Abase + (unsigned long long)tt * TILE_K * 4;
      tdm_load_2d(sa, ga, TILE_K, TILE_M, TILE_K, TILE_M, K_DIM);      // A: [128 rows][32 k]
      unsigned sb = sa + STAGE_A;
      unsigned long long gb = (unsigned long long)Bbase
                            + (unsigned long long)tt * TILE_K * N_DIM * 4;
      tdm_load_2d(sb, gb, TILE_N, TILE_K, TILE_N, TILE_K, N_DIM);      // B: [32 k][128 n]
    }
#else
    float* sA = (float*)(lds_stage + (tt & 1) * STAGE_BUF);
    float* sB = sA + TILE_M * TILE_K;
    #pragma unroll
    for (int i = 0; i < 4; ++i) {
      int f4 = tid * 4 + i; int row = f4 >> 3; int c4 = (f4 & 7) * 4;
      *(v4f*)&sA[row * TILE_K + c4] =
          *(const v4f*)(Abase + (long long)row * K_DIM + tt * TILE_K + c4);
    }
    #pragma unroll
    for (int i = 0; i < 4; ++i) {
      int f4 = tid * 4 + i; int kr = f4 >> 5; int c4 = (f4 & 31) * 4;
      *(v4f*)&sB[kr * TILE_N + c4] =
          *(const v4f*)(Bbase + (long long)(tt * TILE_K + kr) * N_DIM + c4);
    }
#endif
  };

  // ---- convert tile tt: stage[tt&1] f32 -> planes[tt&1] bf16 hi/lo ----
  auto convert_tile = [&](int tt) {
    const float* sA = (const float*)(lds_stage + (tt & 1) * STAGE_BUF);   // [128][32]
    const float* sB = sA + TILE_M * TILE_K;                               // [32][128]
    bf16_t* Ah = (bf16_t*)(lds_planes + (tt & 1) * BUF_B);
    bf16_t* Al = Ah + 1 * PLANE_E;
    bf16_t* Bh = Ah + 2 * PLANE_E;
    bf16_t* Bl = Ah + 3 * PLANE_E;
    #pragma unroll
    for (int i = 0; i < 4; ++i) {
      int f4 = tid * 4 + i; int row = f4 >> 3; int c4 = (f4 & 7) * 4;
      v4f v = *(const v4f*)&sA[row * TILE_K + c4];
      #pragma unroll
      for (int j = 0; j < 4; ++j) {
        bf16_t h, l; split_bf16(v[j], h, l);
        Ah[row * LDSTR + c4 + j] = h;
        Al[row * LDSTR + c4 + j] = l;
      }
    }
    #pragma unroll
    for (int i = 0; i < 4; ++i) {
      int f4 = tid * 4 + i; int kr = f4 >> 5; int c4 = (f4 & 31) * 4;
      v4f v = *(const v4f*)&sB[kr * TILE_N + c4];
      #pragma unroll
      for (int j = 0; j < 4; ++j) {             // transposed: [n][k]
        bf16_t h, l; split_bf16(v[j], h, l);
        Bh[(c4 + j) * LDSTR + kr] = h;
        Bl[(c4 + j) * LDSTR + kr] = l;
      }
    }
  };

  // ---- prologue: tiles 0 and 1 in flight; convert tile 0 ----
  fetch_tile(0);
  fetch_tile(1);
  wait_fetch<2>(wave);    // tile 0's two TDM ops complete (in-order), tile 1 may fly
  __syncthreads();
  convert_tile(0);
  wait_fetch<0>(wave);    // tile 1 staged
  __syncthreads();

  for (int i = 0; i < NTILES; ++i) {
    // TDM for tile i+2 runs under this iteration's WMMA + convert
    if (i + 2 < NTILES) fetch_tile(i + 2);

    bf16_t* Ah = (bf16_t*)(lds_planes + (i & 1) * BUF_B);
    bf16_t* Al = Ah + 1 * PLANE_E;
    bf16_t* Bh = Ah + 2 * PLANE_E;
    bf16_t* Bl = Ah + 3 * PLANE_E;

    // fragments (ISA 16-bit A 16x32 / B 32x16 lane layouts)
    v16bf ah[2], al[2];
    #pragma unroll
    for (int m2 = 0; m2 < 2; ++m2) {
      int m  = wm * 32 + m2 * 16 + lhalf;
      int k8 = lhi * 8;
      ah[m2] = cat8(*(const v8bf*)&Ah[m * LDSTR + k8],
                    *(const v8bf*)&Ah[m * LDSTR + 16 + k8]);
      al[m2] = cat8(*(const v8bf*)&Al[m * LDSTR + k8],
                    *(const v8bf*)&Al[m * LDSTR + 16 + k8]);
    }
    v16bf bh[4], bl[4];
    #pragma unroll
    for (int n2 = 0; n2 < 4; ++n2) {
      int n  = wn * 64 + n2 * 16 + lhalf;
      int kb = lhi * 8;
      bh[n2] = cat8(*(const v8bf*)&Bh[n * LDSTR + kb],
                    *(const v8bf*)&Bh[n * LDSTR + 16 + kb]);
      bl[n2] = cat8(*(const v8bf*)&Bl[n * LDSTR + kb],
                    *(const v8bf*)&Bl[n * LDSTR + 16 + kb]);
    }

    // bf16x3: hi*hi + hi*lo + lo*hi, f32 accumulate
    #pragma unroll
    for (int m2 = 0; m2 < 2; ++m2) {
      #pragma unroll
      for (int n2 = 0; n2 < 4; ++n2) {
        acc[m2][n2] = __builtin_amdgcn_wmma_f32_16x16x32_bf16(
            false, ah[m2], false, bh[n2], (short)0, acc[m2][n2], false, false);
        acc[m2][n2] = __builtin_amdgcn_wmma_f32_16x16x32_bf16(
            false, ah[m2], false, bl[n2], (short)0, acc[m2][n2], false, false);
        acc[m2][n2] = __builtin_amdgcn_wmma_f32_16x16x32_bf16(
            false, al[m2], false, bh[n2], (short)0, acc[m2][n2], false, false);
      }
    }

    // convert tile i+1 (staged last iteration) into the other plane buffer
    if (i + 1 < NTILES) convert_tile(i + 1);

    wait_fetch<0>(wave);  // tile i+2 staged before anyone reads it next iteration
    __syncthreads();
  }

  // ---- epilogue: D 16x16 f32 layout -> row-major C ----
  #pragma unroll
  for (int m2 = 0; m2 < 2; ++m2) {
    #pragma unroll
    for (int n2 = 0; n2 < 4; ++n2) {
      int row0 = wm * 32 + m2 * 16 + lhi * 8;
      int col  = wn * 64 + n2 * 16 + lhalf;
      #pragma unroll
      for (int r = 0; r < 8; ++r) {
        Cbase[(long long)(row0 + r) * N_DIM + col] = acc[m2][n2][r];
      }
    }
  }
}

extern "C" void kernel_launch(void* const* d_in, const int* in_sizes, int n_in,
                              void* d_out, int out_size, void* d_ws, size_t ws_size,
                              hipStream_t stream) {
  const float* x   = (const float*)d_in[0];
  const int*   ids = (const int*)d_in[1];
  const float* w   = (const float*)d_in[2];
  float*       out = (float*)d_out;
  hipLaunchKernelGGL(multilora_gemm_bf16x3, dim3(2048), dim3(256), 0, stream,
                     x, ids, w, out);
}
